// Network_4698694222262
// MI455X (gfx1250) — compile-verified, matrix operand order checked by reading, not run on hardware
//
#include <hip/hip_runtime.h>
#include <hip/hip_bf16.h>

// ---------------------------------------------------------------------------
// CDNA5 (gfx1250) fused equivariant-GNN forward.
// wave32, WMMA f32_16x16x32_f16 for all big GEMMs (radial MLPs + node linear).
// ---------------------------------------------------------------------------

typedef __attribute__((ext_vector_type(16))) _Float16 v16h;
typedef __attribute__((ext_vector_type(8)))  _Float16 v8h;
typedef __attribute__((ext_vector_type(8)))  float    v8f;

#define C_S 0.38268343f   /* sin(pi/8) */
#define C_X 0.92387953f   /* cos(pi/8) */
#define SILU_NORM 1.6791767f
#define INV_SQRT8  0.35355339f
#define INV_8      0.125f
#define INV_SQRT32 0.17677669f
#define INV_SQRT3  0.57735027f
#define SCAT       0.25f  /* 1/sqrt(16) */

__device__ __forceinline__ float silu_n(float x) {
    return SILU_NORM * x / (1.0f + __expf(-x));
}

__device__ __forceinline__ float atomic_fadd(float* p, float v) {
    return __hip_atomic_fetch_add(p, v, __ATOMIC_RELAXED, __HIP_MEMORY_SCOPE_AGENT);
}

// Load one 16x16x32 f16 fragment (A or B) from a row-major [rows][Kpad] f16
// LDS tile.  Per ISA 7.12.2: lane<16 holds K runs {0..7} and {16..23},
// lane>=16 holds {8..15} and {24..31}, row = lane&15 (M for A, N for B^T).
// `rowbase` must already point at (row, kb).
__device__ __forceinline__ v16h frag_ld(const _Float16* rowbase, int lane) {
    const _Float16* p = rowbase + ((lane & 16) ? 8 : 0);
    v8h lo = *(const v8h*)(p);
    v8h hi = *(const v8h*)(p + 16);
    v16h f;
#pragma unroll
    for (int i = 0; i < 8; ++i) { f[i] = lo[i]; f[i + 8] = hi[i]; }
    return f;
}

__device__ __forceinline__ v8f wmma16(v16h a, v16h b, v8f c) {
    return __builtin_amdgcn_wmma_f32_16x16x32_f16(
        false, a, false, b, (short)0, c, false, false);
}

// Scatter a D fragment back into a row-major [16][Kpad] f16 LDS tile with
// scale + optional normalized-SiLU.  D layout: VGPR r -> M=r(+8), N=lane&15.
__device__ __forceinline__ void frag_st_h(_Float16* base, int Kpad, v8f c,
                                          int col0, int lane, float scale,
                                          bool do_silu) {
    int n  = col0 + (lane & 15);
    int mb = (lane & 16) ? 8 : 0;
#pragma unroll
    for (int r = 0; r < 8; ++r) {
        float v = c[r] * scale;
        if (do_silu) v = silu_n(v);
        base[(r + mb) * Kpad + n] = (_Float16)v;
    }
}

// ---------------------------------------------------------------------------
// K1: per-node pre-linears:  [sc_sca | sc_gat | h0] = node_fea @ Wc / 8
// One wave per 16-node tile, 8 column tiles, K=64.
// ---------------------------------------------------------------------------
__global__ __launch_bounds__(192) void node_pre_kernel(
    const float* __restrict__ fea, const float* __restrict__ Wsca,
    const float* __restrict__ Wgat, const float* __restrict__ Wlin,
    float* __restrict__ sc_sca, float* __restrict__ sc_gat,
    float* __restrict__ h0, int N, int ntiles)
{
    __shared__ _Float16 sWt[128 * 64];   // [outCol][inK] transposed f16
    __shared__ _Float16 sA[6][16 * 64];

    const int tid = threadIdx.x, nt = blockDim.x;
    for (int i = tid; i < 8192; i += nt) {
        int c = i >> 6, u = i & 63;
        float w = (c < 32) ? Wsca[u * 32 + c]
                : (c < 64) ? Wgat[u * 32 + (c - 32)]
                           : Wlin[u * 64 + (c - 64)];
        sWt[i] = (_Float16)w;
    }
    __syncthreads();

    const int wv = tid >> 5, lane = tid & 31, wpb = nt >> 5;
    for (int tile = blockIdx.x * wpb + wv; tile < ntiles;
         tile += gridDim.x * wpb) {
        {   // stage 16x64 activation tile as f16
            int m = lane & 15, c0 = (lane & 16) ? 32 : 0;
            int node = tile * 16 + m;
#pragma unroll
            for (int c = 0; c < 32; ++c)
                sA[wv][m * 64 + c0 + c] =
                    (node < N) ? (_Float16)fea[node * 64 + c0 + c]
                               : (_Float16)0.0f;
        }
        v8f acc[8] = {};
#pragma unroll
        for (int kb = 0; kb < 64; kb += 32) {
            v16h a = frag_ld(&sA[wv][(lane & 15) * 64 + kb], lane);
#pragma unroll
            for (int t = 0; t < 8; ++t) {
                v16h b = frag_ld(&sWt[(t * 16 + (lane & 15)) * 64 + kb], lane);
                acc[t] = wmma16(a, b, acc[t]);
            }
        }
        int cl = lane & 15, mb = (lane & 16) ? 8 : 0;
#pragma unroll
        for (int t = 0; t < 8; ++t) {
            int col = t * 16 + cl;
#pragma unroll
            for (int r = 0; r < 8; ++r) {
                int node = tile * 16 + r + mb;
                if (node < N) {
                    float v = acc[t][r] * INV_8;
                    if (col < 32)      sc_sca[node * 32 + col] = v;
                    else if (col < 64) sc_gat[node * 32 + col - 32] = v;
                    else               h0[node * 64 + col - 64] = v;
                }
            }
        }
    }
}

// ---------------------------------------------------------------------------
// K2: fused layer-1 edge kernel.  Per 16-edge wave tile:
//   geometry -> bessel emb -> radial MLP (8->64->64->128, WMMA) ->
//   gather h0[src] -> m0/m1 messages -> atomic scatter into a0 (N,64) and
//   a1 (N,64,3), all scaled by 1/sqrt(16).
// ---------------------------------------------------------------------------
__global__ __launch_bounds__(192) void edge_l1_kernel(
    const float* __restrict__ evec, const int* __restrict__ esrc,
    const int* __restrict__ edst, const float* __restrict__ fc0,
    const float* __restrict__ fc1, const float* __restrict__ fc2,
    const float* __restrict__ h0g, float* __restrict__ a0,
    float* __restrict__ a1, int E, int ntiles)
{
    __shared__ float    sW0[8 * 64];        // layer0 weights, f32
    __shared__ _Float16 sWt1[64 * 64];      // [out][in] transposed
    __shared__ _Float16 sWt2[128 * 64];     // [out][in] transposed
    __shared__ float    sEmb[6][16 * 8];
    __shared__ _Float16 sX1[6][16 * 64];
    __shared__ _Float16 sX2[6][16 * 64];
    __shared__ float    sY1[6][16 * 3];
    __shared__ int      sSrc[6][16];
    __shared__ int      sDst[6][16];

    const int tid = threadIdx.x, nt = blockDim.x;
    for (int i = tid; i < 512; i += nt) sW0[i] = fc0[i];
    for (int i = tid; i < 4096; i += nt) {
        int v = i >> 6, u = i & 63;
        sWt1[i] = (_Float16)fc1[u * 64 + v];
    }
    for (int i = tid; i < 8192; i += nt) {
        int c = i >> 6, u = i & 63;
        sWt2[i] = (_Float16)fc2[u * 128 + c];
    }
    __syncthreads();

    const int wv = tid >> 5, lane = tid & 31, wpb = nt >> 5;
    for (int tile = blockIdx.x * wpb + wv; tile < ntiles;
         tile += gridDim.x * wpb) {
        // --- edge geometry + bessel basis (lanes 0..15, one edge each) ---
        if (lane < 16) {
            int e = tile * 16 + lane;
            float vx = 0.f, vy = 0.f, vz = 0.f; int s = 0, d = 0;
            if (e < E) {
                vx = evec[e * 3]; vy = evec[e * 3 + 1]; vz = evec[e * 3 + 2];
                s = esrc[e]; d = edst[e];
            }
            float r = sqrtf(vx * vx + vy * vy + vz * vz + 1e-12f);
            float x = r * 0.25f;               // r / R_MAX
            float cut = 0.0f;
            if (e < E && x < 1.0f) {
                float x2 = x * x, x3 = x2 * x, x6 = x3 * x3;
                cut = 1.0f - 28.0f * x6 + 48.0f * x6 * x - 21.0f * x6 * x2;
            }
            float invr = 1.0f / r;
            float pref = 0.5f * invr * cut;    // (2/R_MAX) * cut / r
#pragma unroll
            for (int k = 0; k < 8; ++k)
                sEmb[wv][lane * 8 + k] =
                    pref * __sinf(x * (3.14159265f * (float)(k + 1)));
            float ys = (e < E) ? 1.7320508f * invr : 0.0f;  // sqrt(3)/r
            sY1[wv][lane * 3 + 0] = ys * vx;
            sY1[wv][lane * 3 + 1] = ys * vy;
            sY1[wv][lane * 3 + 2] = ys * vz;
            sSrc[wv][lane] = s; sDst[wv][lane] = d;
        }
        // --- layer0: K=8, VALU (too thin for WMMA). lane -> (row, 32 cols) -
        {
            int m = lane & 15, c0 = (lane & 16) ? 32 : 0;
            float ev[8];
#pragma unroll
            for (int k = 0; k < 8; ++k) ev[k] = sEmb[wv][m * 8 + k];
#pragma unroll
            for (int c = 0; c < 32; ++c) {
                float s = 0.f;
#pragma unroll
                for (int k = 0; k < 8; ++k) s += ev[k] * sW0[k * 64 + c0 + c];
                sX1[wv][m * 64 + c0 + c] = (_Float16)silu_n(s * INV_SQRT8);
            }
        }
        // --- layer1: 16x64x64 WMMA, X1 -> X2 (silu, /8) ---
        {
            v8f acc[4] = {};
#pragma unroll
            for (int kb = 0; kb < 64; kb += 32) {
                v16h a = frag_ld(&sX1[wv][(lane & 15) * 64 + kb], lane);
#pragma unroll
                for (int t = 0; t < 4; ++t) {
                    v16h b = frag_ld(&sWt1[(t * 16 + (lane & 15)) * 64 + kb],
                                     lane);
                    acc[t] = wmma16(a, b, acc[t]);
                }
            }
#pragma unroll
            for (int t = 0; t < 4; ++t)
                frag_st_h(sX2[wv], 64, acc[t], t * 16, lane, INV_8, true);
        }
        // --- layer2: 16x128x64 WMMA, keep w in registers ---
        v8f wacc[8] = {};
#pragma unroll
        for (int kb = 0; kb < 64; kb += 32) {
            v16h a = frag_ld(&sX2[wv][(lane & 15) * 64 + kb], lane);
#pragma unroll
            for (int t = 0; t < 8; ++t) {
                v16h b = frag_ld(&sWt2[(t * 16 + (lane & 15)) * 64 + kb],
                                 lane);
                wacc[t] = wmma16(a, b, wacc[t]);
            }
        }
        // --- messages + atomic scatter ---
        {
            int cl = lane & 15, mb = (lane & 16) ? 8 : 0;
#pragma unroll
            for (int t = 0; t < 4; ++t) {        // w cols 0..63 -> m0
                int col = t * 16 + cl;
#pragma unroll
                for (int r = 0; r < 8; ++r) {
                    int m = r + mb;
                    int src = sSrc[wv][m], dst = sDst[wv][m];
                    float wvv = wacc[t][r] * INV_8;
                    float val = wvv * h0g[src * 64 + col] * SCAT;
                    atomic_fadd(&a0[dst * 64 + col], val);
                }
            }
#pragma unroll
            for (int t = 4; t < 8; ++t) {        // w cols 64..127 -> m1 (x y1)
                int c = (t - 4) * 16 + cl;
#pragma unroll
                for (int r = 0; r < 8; ++r) {
                    int m = r + mb;
                    int src = sSrc[wv][m], dst = sDst[wv][m];
                    float wvv = wacc[t][r] * INV_8;
                    float b = wvv * h0g[src * 64 + c] * SCAT;
                    float* p = &a1[(dst * 64 + c) * 3];
                    atomic_fadd(p + 0, b * sY1[wv][m * 3 + 0]);
                    atomic_fadd(p + 1, b * sY1[wv][m * 3 + 1]);
                    atomic_fadd(p + 2, b * sY1[wv][m * 3 + 2]);
                }
            }
        }
    }
}

// ---------------------------------------------------------------------------
// K3a: per-node mid stage 1: sca/gat/v1 -> f0 (N,32), f1 (N,32,3)
// ---------------------------------------------------------------------------
__global__ void node_mid1_kernel(
    const float* __restrict__ a0, const float* __restrict__ a1,
    const float* __restrict__ sc_sca, const float* __restrict__ sc_gat,
    const float* __restrict__ Wsca2, const float* __restrict__ Wgat2,
    const float* __restrict__ W1o, float* __restrict__ f0g,
    float* __restrict__ f1g, int N)
{
    int t = blockIdx.x * blockDim.x + threadIdx.x;
    if (t >= N * 32) return;
    int n = t >> 5, v = t & 31;
    float s = 0.f, g = 0.f;
    for (int u = 0; u < 64; ++u) {
        float a = a0[n * 64 + u];
        s += a * Wsca2[u * 32 + v];
        g += a * Wgat2[u * 32 + v];
    }
    s = C_S * sc_sca[n * 32 + v] + C_X * s * INV_8;
    g = C_S * sc_gat[n * 32 + v] + C_X * g * INV_8;
    f0g[n * 32 + v] = silu_n(s);
    float gact = silu_n(g);
#pragma unroll
    for (int d = 0; d < 3; ++d) {
        float acc = 0.f;
        for (int u = 0; u < 64; ++u)
            acc += a1[(n * 64 + u) * 3 + d] * W1o[u * 32 + v];
        f1g[(n * 32 + v) * 3 + d] = acc * INV_8 * gact;
    }
}

// ---------------------------------------------------------------------------
// K3b: per-node mid stage 2: sc_sca2 (N,32), h0_2 (N,32), h1_2 (N,32,3)
// (l2_sc_1o / sc_v1 are dead code in the reference and skipped.)
// ---------------------------------------------------------------------------
__global__ void node_mid2_kernel(
    const float* __restrict__ f0g, const float* __restrict__ f1g,
    const float* __restrict__ Wsc0e, const float* __restrict__ Wlin0e,
    const float* __restrict__ Wlin1o, float* __restrict__ sc2,
    float* __restrict__ h0_2, float* __restrict__ h1_2, int N)
{
    int t = blockIdx.x * blockDim.x + threadIdx.x;
    if (t >= N * 32) return;
    int n = t >> 5, v = t & 31;
    float ss = 0.f, hh = 0.f;
    for (int u = 0; u < 32; ++u) {
        float f = f0g[n * 32 + u];
        ss += f * Wsc0e[u * 96 + v];      // l2_sc_0e is (32,96), cols [0,32)
        hh += f * Wlin0e[u * 32 + v];
    }
    sc2[n * 32 + v]  = ss * INV_SQRT32;
    h0_2[n * 32 + v] = hh * INV_SQRT32;
#pragma unroll
    for (int d = 0; d < 3; ++d) {
        float acc = 0.f;
        for (int u = 0; u < 32; ++u)
            acc += f1g[(n * 32 + u) * 3 + d] * Wlin1o[u * 32 + v];
        h1_2[(n * 32 + v) * 3 + d] = acc * INV_SQRT32;
    }
}

// ---------------------------------------------------------------------------
// K4: fused layer-2 edge kernel.  Only the live part of m_0e survives dead-
// code elimination: w cols [0:32) * s0 and w cols [96:128) * (s1.y1)/sqrt(3),
// scattered into a0_2 (N,64).  l2_fc W2 is sliced to those 64 columns.
// ---------------------------------------------------------------------------
__global__ __launch_bounds__(192) void edge_l2_kernel(
    const float* __restrict__ evec, const int* __restrict__ esrc,
    const int* __restrict__ edst, const float* __restrict__ fc0,
    const float* __restrict__ fc1, const float* __restrict__ fc2,
    const float* __restrict__ h0g, const float* __restrict__ h1g,
    float* __restrict__ a0_2, int E, int ntiles)
{
    __shared__ float    sW0[8 * 64];
    __shared__ _Float16 sWt1[64 * 64];
    __shared__ _Float16 sWt2[64 * 64];   // sliced cols [0:32)U[96:128)
    __shared__ float    sEmb[6][16 * 8];
    __shared__ _Float16 sX1[6][16 * 64];
    __shared__ _Float16 sX2[6][16 * 64];
    __shared__ float    sY1[6][16 * 3];
    __shared__ int      sSrc[6][16];
    __shared__ int      sDst[6][16];

    const int tid = threadIdx.x, nt = blockDim.x;
    for (int i = tid; i < 512; i += nt) sW0[i] = fc0[i];
    for (int i = tid; i < 4096; i += nt) {
        int v = i >> 6, u = i & 63;
        sWt1[i] = (_Float16)fc1[u * 64 + v];
    }
    for (int i = tid; i < 4096; i += nt) {
        int c = i >> 6, u = i & 63;
        int oc = (c < 32) ? c : (c + 64);   // 0..31 and 96..127 of 160
        sWt2[i] = (_Float16)fc2[u * 160 + oc];
    }
    __syncthreads();

    const int wv = tid >> 5, lane = tid & 31, wpb = nt >> 5;
    for (int tile = blockIdx.x * wpb + wv; tile < ntiles;
         tile += gridDim.x * wpb) {
        if (lane < 16) {
            int e = tile * 16 + lane;
            float vx = 0.f, vy = 0.f, vz = 0.f; int s = 0, d = 0;
            if (e < E) {
                vx = evec[e * 3]; vy = evec[e * 3 + 1]; vz = evec[e * 3 + 2];
                s = esrc[e]; d = edst[e];
            }
            float r = sqrtf(vx * vx + vy * vy + vz * vz + 1e-12f);
            float x = r * 0.25f;
            float cut = 0.0f;
            if (e < E && x < 1.0f) {
                float x2 = x * x, x3 = x2 * x, x6 = x3 * x3;
                cut = 1.0f - 28.0f * x6 + 48.0f * x6 * x - 21.0f * x6 * x2;
            }
            float invr = 1.0f / r;
            float pref = 0.5f * invr * cut;
#pragma unroll
            for (int k = 0; k < 8; ++k)
                sEmb[wv][lane * 8 + k] =
                    pref * __sinf(x * (3.14159265f * (float)(k + 1)));
            float ys = (e < E) ? 1.7320508f * invr : 0.0f;
            sY1[wv][lane * 3 + 0] = ys * vx;
            sY1[wv][lane * 3 + 1] = ys * vy;
            sY1[wv][lane * 3 + 2] = ys * vz;
            sSrc[wv][lane] = s; sDst[wv][lane] = d;
        }
        {
            int m = lane & 15, c0 = (lane & 16) ? 32 : 0;
            float ev[8];
#pragma unroll
            for (int k = 0; k < 8; ++k) ev[k] = sEmb[wv][m * 8 + k];
#pragma unroll
            for (int c = 0; c < 32; ++c) {
                float s = 0.f;
#pragma unroll
                for (int k = 0; k < 8; ++k) s += ev[k] * sW0[k * 64 + c0 + c];
                sX1[wv][m * 64 + c0 + c] = (_Float16)silu_n(s * INV_SQRT8);
            }
        }
        {
            v8f acc[4] = {};
#pragma unroll
            for (int kb = 0; kb < 64; kb += 32) {
                v16h a = frag_ld(&sX1[wv][(lane & 15) * 64 + kb], lane);
#pragma unroll
                for (int t = 0; t < 4; ++t) {
                    v16h b = frag_ld(&sWt1[(t * 16 + (lane & 15)) * 64 + kb],
                                     lane);
                    acc[t] = wmma16(a, b, acc[t]);
                }
            }
#pragma unroll
            for (int t = 0; t < 4; ++t)
                frag_st_h(sX2[wv], 64, acc[t], t * 16, lane, INV_8, true);
        }
        v8f wacc[4] = {};
#pragma unroll
        for (int kb = 0; kb < 64; kb += 32) {
            v16h a = frag_ld(&sX2[wv][(lane & 15) * 64 + kb], lane);
#pragma unroll
            for (int t = 0; t < 4; ++t) {
                v16h b = frag_ld(&sWt2[(t * 16 + (lane & 15)) * 64 + kb],
                                 lane);
                wacc[t] = wmma16(a, b, wacc[t]);
            }
        }
        {
            int cl = lane & 15, mb = (lane & 16) ? 8 : 0;
#pragma unroll
            for (int t = 0; t < 2; ++t) {        // w cols 0..31: * s0
                int col = t * 16 + cl;
#pragma unroll
                for (int r = 0; r < 8; ++r) {
                    int m = r + mb;
                    int src = sSrc[wv][m], dst = sDst[wv][m];
                    float wvv = wacc[t][r] * INV_8;
                    float val = wvv * h0g[src * 32 + col] * SCAT;
                    atomic_fadd(&a0_2[dst * 64 + col], val);
                }
            }
#pragma unroll
            for (int t = 2; t < 4; ++t) {        // w cols 96..127: * (s1.y1)/sqrt3
                int c = (t - 2) * 16 + cl;
#pragma unroll
                for (int r = 0; r < 8; ++r) {
                    int m = r + mb;
                    int src = sSrc[wv][m], dst = sDst[wv][m];
                    float wvv = wacc[t][r] * INV_8;
                    const float* s1 = &h1g[(src * 32 + c) * 3];
                    float sdot = s1[0] * sY1[wv][m * 3 + 0]
                               + s1[1] * sY1[wv][m * 3 + 1]
                               + s1[2] * sY1[wv][m * 3 + 2];
                    float val = wvv * sdot * (INV_SQRT3 * SCAT);
                    atomic_fadd(&a0_2[dst * 64 + 32 + c], val);
                }
            }
        }
    }
}

// ---------------------------------------------------------------------------
// K5: final per-node mix: f0_2 = silu(C_S*sc2 + C_X*(a0_2 @ l2_lin2_0e[:,:32])/8)
// ---------------------------------------------------------------------------
__global__ void node_fin_kernel(
    const float* __restrict__ a0_2, const float* __restrict__ sc2,
    const float* __restrict__ W0e2, float* __restrict__ f0_2, int N)
{
    int t = blockIdx.x * blockDim.x + threadIdx.x;
    if (t >= N * 32) return;
    int n = t >> 5, v = t & 31;
    float t2 = 0.f;
    for (int u = 0; u < 64; ++u)
        t2 += a0_2[n * 64 + u] * W0e2[u * 96 + v];   // (64,96), cols [0,32)
    float sca = C_S * sc2[n * 32 + v] + C_X * t2 * INV_8;
    f0_2[n * 32 + v] = silu_n(sca);
}

// ---------------------------------------------------------------------------
// K6: out[n] = ((f0_2 @ lin1)/sqrt32 @ lin2)/sqrt32, with lin1@lin2 pre-folded
// ---------------------------------------------------------------------------
__global__ void node_out_kernel(
    const float* __restrict__ f0_2, const float* __restrict__ lin1,
    const float* __restrict__ lin2, float* __restrict__ out, int N)
{
    __shared__ float cvec[32];
    int tid = threadIdx.x;
    if (tid < 32) {
        float c = 0.f;
        for (int v = 0; v < 32; ++v) c += lin1[tid * 32 + v] * lin2[v];
        cvec[tid] = c * (INV_SQRT32 * INV_SQRT32);   // 1/32
    }
    __syncthreads();
    int n = blockIdx.x * blockDim.x + tid;
    if (n < N) {
        float o = 0.f;
        for (int u = 0; u < 32; ++u) o += f0_2[n * 32 + u] * cvec[u];
        out[n] = o;
    }
}

// ---------------------------------------------------------------------------
extern "C" void kernel_launch(void* const* d_in, const int* in_sizes, int n_in,
                              void* d_out, int out_size, void* d_ws,
                              size_t ws_size, hipStream_t stream) {
    (void)n_in; (void)out_size; (void)ws_size;
    const float* node_fea  = (const float*)d_in[0];
    const int*   edge_idx  = (const int*)d_in[1];
    const float* edge_vec  = (const float*)d_in[2];
    const int N = in_sizes[0] / 64;
    const int E = in_sizes[2] / 3;
    const int* esrc = edge_idx;
    const int* edst = edge_idx + E;

    // Disambiguate params flattening: insertion order (in_sizes[4]==2048,
    // l1_sc_sca) vs pytree alphabetical order (in_sizes[4]==512, l1_fc[0]).
    const bool alpha = (in_sizes[4] == 512);
    const float *l1_sc_sca, *l1_sc_gat, *l1_lin1, *l1_fc0, *l1_fc1, *l1_fc2;
    const float *l1_lin2_sca, *l1_lin2_gat, *l1_lin2_1o;
    const float *l2_sc_0e, *l2_lin1_0e, *l2_lin1_1o;
    const float *l2_fc0, *l2_fc1, *l2_fc2, *l2_lin2_0e, *lin1, *lin2;
    if (alpha) {
        l1_fc0 = (const float*)d_in[4];  l1_fc1 = (const float*)d_in[5];
        l1_fc2 = (const float*)d_in[6];  l1_lin1 = (const float*)d_in[7];
        l1_lin2_1o = (const float*)d_in[8];  l1_lin2_gat = (const float*)d_in[9];
        l1_lin2_sca = (const float*)d_in[10]; l1_sc_gat = (const float*)d_in[11];
        l1_sc_sca = (const float*)d_in[12];
        l2_fc0 = (const float*)d_in[13]; l2_fc1 = (const float*)d_in[14];
        l2_fc2 = (const float*)d_in[15];
        l2_lin1_0e = (const float*)d_in[16]; l2_lin1_1o = (const float*)d_in[17];
        l2_lin2_0e = (const float*)d_in[18];
        l2_sc_0e = (const float*)d_in[21];
        lin1 = (const float*)d_in[23]; lin2 = (const float*)d_in[24];
    } else {
        l1_sc_sca = (const float*)d_in[4];  l1_sc_gat = (const float*)d_in[5];
        l1_lin1 = (const float*)d_in[6];
        l1_fc0 = (const float*)d_in[7]; l1_fc1 = (const float*)d_in[8];
        l1_fc2 = (const float*)d_in[9];
        l1_lin2_sca = (const float*)d_in[10]; l1_lin2_gat = (const float*)d_in[11];
        l1_lin2_1o = (const float*)d_in[12];
        l2_sc_0e = (const float*)d_in[13];
        l2_lin1_0e = (const float*)d_in[15]; l2_lin1_1o = (const float*)d_in[16];
        l2_fc0 = (const float*)d_in[17]; l2_fc1 = (const float*)d_in[18];
        l2_fc2 = (const float*)d_in[19];
        l2_lin2_0e = (const float*)d_in[20];
        lin1 = (const float*)d_in[23]; lin2 = (const float*)d_in[24];
    }

    // workspace layout (f32, 256B aligned slices)
    char* ws = (char*)d_ws;
    size_t off = 0;
    auto take = [&](size_t nelem) -> float* {
        float* p = (float*)(ws + off);
        off += (nelem * sizeof(float) + 255) & ~(size_t)255;
        return p;
    };
    float* h0     = take((size_t)N * 64);
    float* sc_sca = take((size_t)N * 32);
    float* sc_gat = take((size_t)N * 32);
    float* a0     = take((size_t)N * 64);
    float* a1     = take((size_t)N * 64 * 3);
    float* f0g    = take((size_t)N * 32);
    float* f1g    = take((size_t)N * 32 * 3);
    float* sc2    = take((size_t)N * 32);
    float* h0_2   = take((size_t)N * 32);
    float* h1_2   = take((size_t)N * 32 * 3);
    float* a0_2   = take((size_t)N * 64);
    float* f0_2   = take((size_t)N * 32);

    const int ntilesN = (N + 15) / 16;
    const int ntilesE = (E + 15) / 16;

    hipMemsetAsync(a0, 0, (size_t)N * 64 * sizeof(float), stream);
    hipMemsetAsync(a1, 0, (size_t)N * 64 * 3 * sizeof(float), stream);
    hipMemsetAsync(a0_2, 0, (size_t)N * 64 * sizeof(float), stream);

    node_pre_kernel<<<(ntilesN + 5) / 6, 192, 0, stream>>>(
        node_fea, l1_sc_sca, l1_sc_gat, l1_lin1, sc_sca, sc_gat, h0, N,
        ntilesN);

    edge_l1_kernel<<<(ntilesE + 5) / 6, 192, 0, stream>>>(
        edge_vec, esrc, edst, l1_fc0, l1_fc1, l1_fc2, h0, a0, a1, E, ntilesE);

    node_mid1_kernel<<<(N * 32 + 255) / 256, 256, 0, stream>>>(
        a0, a1, sc_sca, sc_gat, l1_lin2_sca, l1_lin2_gat, l1_lin2_1o, f0g,
        f1g, N);

    node_mid2_kernel<<<(N * 32 + 255) / 256, 256, 0, stream>>>(
        f0g, f1g, l2_sc_0e, l2_lin1_0e, l2_lin1_1o, sc2, h0_2, h1_2, N);

    edge_l2_kernel<<<(ntilesE + 5) / 6, 192, 0, stream>>>(
        edge_vec, esrc, edst, l2_fc0, l2_fc1, l2_fc2, h0_2, h1_2, a0_2, E,
        ntilesE);

    node_fin_kernel<<<(N * 32 + 255) / 256, 256, 0, stream>>>(
        a0_2, sc2, l2_lin2_0e, f0_2, N);

    node_out_kernel<<<(N + 255) / 256, 256, 0, stream>>>(
        f0_2, lin1, lin2, (float*)d_out, N);
}
